// GroupedQueryAttention_67173288509677
// MI455X (gfx1250) — matrix operand order repaired and checked by reading, not compiled, not run on hardware
//
#include <hip/hip_runtime.h>
#include <hip/hip_bf16.h>

typedef __bf16 bf16_t;
typedef __attribute__((ext_vector_type(16))) __bf16 v16bf;
typedef __attribute__((ext_vector_type(8)))  __bf16 v8bf;
typedef __attribute__((ext_vector_type(8)))  float  v8f;

#define D_MODEL 1024
#define NH      16
#define DHD     64
#define NGROUP  8
#define LGRP    512
#define NBATCH  4
#define SEQLEN  4096
#define NROWS   (NBATCH * SEQLEN)   // 16384
#define SSTR    516                 // score row stride (f32), 512+4 pad

// LDS byte offsets for the attention kernel
#define OFF_RINV  (64 * SSTR * 4)              // 132096
#define OFF_KSH   (OFF_RINV + 256)             // 132352
#define KSTR      72                           // K tile row stride (bf16)
#define OFF_VSH   (OFF_KSH + 512 * KSTR * 2)   // 206080
#define VSTR      520                          // V tile row stride (bf16)
#define SMEM_ATTN (OFF_VSH + 64 * VSTR * 2)    // 272640 (< 320KB WGP LDS)

__device__ __forceinline__ v8f wmma_bf16(v16bf a, v16bf b, v8f c) {
  return __builtin_amdgcn_wmma_f32_16x16x32_bf16(
      false, a, false, b, (short)0, c, false, false);
}

// Async bulk copy: 16 B per lane, global -> LDS, tracked by ASYNCcnt.
__device__ __forceinline__ void async_copy_b128(void* lds_ptr, const void* gptr) {
  uint32_t l = (uint32_t)(uintptr_t)lds_ptr;   // low 32 bits = LDS byte address
  asm volatile("global_load_async_to_lds_b128 %0, %1, off"
               :: "v"(l), "v"(gptr) : "memory");
}
__device__ __forceinline__ void wait_async_le32() {
  asm volatile("s_wait_asynccnt 0x20" ::: "memory");
}
__device__ __forceinline__ void wait_async_0() {
  asm volatile("s_wait_asynccnt 0x0" ::: "memory");
}

// ---------------------------------------------------------------------------
// Kernel 1: weights -> bf16
// ---------------------------------------------------------------------------
__global__ void cvt_weights_kernel(const float* __restrict__ wqkv,
                                   const float* __restrict__ wout,
                                   bf16_t* __restrict__ wqkv_h,
                                   bf16_t* __restrict__ wout_h) {
  int i = blockIdx.x * blockDim.x + threadIdx.x;
  if (i < 3 * D_MODEL * D_MODEL) wqkv_h[i] = (bf16_t)wqkv[i];
  if (i < D_MODEL * D_MODEL)     wout_h[i] = (bf16_t)wout[i];
}

// ---------------------------------------------------------------------------
// Kernel 2: fused QKV projection.  Each wave: 16x64 tile, K-loop of 32.
// Q scaled by dh^-0.5 on store; V stored transposed [b,h,g,dh,lg].
// Section routing is a *uniform* branch on ngrp (no exec-mask divergence).
// ---------------------------------------------------------------------------
__global__ void qkv_gemm_kernel(const float* __restrict__ x,
                                const bf16_t* __restrict__ wq,
                                bf16_t* __restrict__ Qh,
                                bf16_t* __restrict__ Kh,
                                bf16_t* __restrict__ Vt) {
  const int NGRP = (3 * D_MODEL) / 64;  // 48
  int wid   = blockIdx.x * (blockDim.x >> 5) + (threadIdx.x >> 5);
  int mtile = wid / NGRP;
  int ngrp  = wid % NGRP;
  int lane  = threadIdx.x & 31;
  int lm = lane & 15, lh = lane >> 4;

  const float* xrow = x + (size_t)(mtile * 16 + lm) * D_MODEL;
  v8f acc[4] = {};

  for (int kk = 0; kk < D_MODEL; kk += 32) {
    __builtin_prefetch(xrow + kk + 256, 0, 0);
    v16bf a;
#pragma unroll
    for (int e = 0; e < 8; ++e) {
      a[e]     = (bf16_t)xrow[kk + lh * 16 + e];
      a[8 + e] = (bf16_t)xrow[kk + lh * 16 + 8 + e];
    }
#pragma unroll
    for (int t = 0; t < 4; ++t) {
      int n = ngrp * 64 + t * 16 + lm;                 // B column n = Wqkv row n
      v16bf bfrag = *(const v16bf*)(wq + (size_t)n * D_MODEL + kk + lh * 16);
      acc[t] = wmma_bf16(a, bfrag, acc[t]);
    }
  }

  int sec  = ngrp >> 4;                 // 0=Q, 1=K, 2=V  (uniform)
  int ncs  = (ngrp & 15) * 64;          // column base within section
  if (sec == 0) {
#pragma unroll
    for (int t = 0; t < 4; ++t) {
      int nc = ncs + t * 16 + lm;
#pragma unroll
      for (int e = 0; e < 8; ++e) {
        int row = mtile * 16 + e + lh * 8;
        Qh[(size_t)row * D_MODEL + nc] = (bf16_t)(acc[t][e] * 0.125f);
      }
    }
  } else if (sec == 1) {
#pragma unroll
    for (int t = 0; t < 4; ++t) {
      int nc = ncs + t * 16 + lm;
#pragma unroll
      for (int e = 0; e < 8; ++e) {
        int row = mtile * 16 + e + lh * 8;
        Kh[(size_t)row * D_MODEL + nc] = (bf16_t)acc[t][e];
      }
    }
  } else {
#pragma unroll
    for (int t = 0; t < 4; ++t) {
      int d = ncs + t * 16 + lm;        // 0..1023 within V
      int h = d >> 6, dd = d & 63;
#pragma unroll
      for (int e = 0; e < 8; ++e) {
        int row = mtile * 16 + e + lh * 8;
        int bb = row >> 12, l = row & 4095;
        int g  = l >> 9,  ll = l & 511;
        Vt[(((size_t)((bb * NH + h) * NGROUP + g) * DHD + dd) * LGRP) + ll] =
            (bf16_t)acc[t][e];
      }
    }
  }
}

// ---------------------------------------------------------------------------
// Kernel 3: block-local attention.  WG = 4 waves / 128 thr, 64 query rows of
// one (b,h,g) block.  K (64KB) and V^T (64KB) tiles staged into LDS with
// GLOBAL_LOAD_ASYNC_TO_LDS_B128; full 64x512 f32 score panel in LDS (132KB).
// ---------------------------------------------------------------------------
__global__ void attn_kernel(const bf16_t* __restrict__ Qh,
                            const bf16_t* __restrict__ Kh,
                            const bf16_t* __restrict__ Vt,
                            bf16_t* __restrict__ Oh) {
  extern __shared__ char smemraw[];
  float*  Ssh  = (float*)smemraw;                 // [64][SSTR] scores
  float*  rInv = (float*)(smemraw + OFF_RINV);    // [64] 1/rowsum
  bf16_t* Ksh  = (bf16_t*)(smemraw + OFF_KSH);    // [512][KSTR]
  bf16_t* Vsh  = (bf16_t*)(smemraw + OFF_VSH);    // [64][VSTR]

  int tid  = threadIdx.x;
  int wave = tid >> 5;
  int lane = tid & 31;
  int lm = lane & 15, lh = lane >> 4;

  int bid  = blockIdx.x;
  int qc   = bid & 7;
  int rest = bid >> 3;
  int g = rest & 7, h = (rest >> 3) & 15, bb = rest >> 7;

  size_t rowbase = (size_t)bb * SEQLEN + (size_t)g * LGRP;
  size_t vbase   = ((size_t)((bb * NH + h) * NGROUP + g)) * DHD * LGRP;

  // ---- Async stage: K block rows (512 x 64 bf16) then V^T (64 x 512) ----
#pragma unroll 4
  for (int i = 0; i < 32; ++i) {              // 32 issues/wave for K
    int idx = i * 128 + tid;                  // 0..4095
    int row = idx >> 3, seg = idx & 7;        // 8 x 16B segs per row
    async_copy_b128(Ksh + row * KSTR + seg * 8,
                    Kh + (rowbase + row) * D_MODEL + h * DHD + seg * 8);
  }
#pragma unroll 4
  for (int i = 0; i < 32; ++i) {              // 32 issues/wave for V
    int idx = i * 128 + tid;
    int row = idx >> 6, seg = idx & 63;       // 64 x 16B segs per dh-row
    async_copy_b128(Vsh + row * VSTR + seg * 8,
                    Vt + vbase + (size_t)row * LGRP + seg * 8);
  }

  // Hoisted Q fragments (loop-invariant over key tiles)
  const bf16_t* qptr =
      Qh + (rowbase + qc * 64 + wave * 16 + lm) * D_MODEL + h * DHD;
  v16bf qa[2];
#pragma unroll
  for (int kc2 = 0; kc2 < 2; ++kc2) {
    v8bf a0 = *(const v8bf*)(qptr + kc2 * 32 + lh * 16);
    v8bf a1 = *(const v8bf*)(qptr + kc2 * 32 + lh * 16 + 8);
#pragma unroll
    for (int e = 0; e < 8; ++e) { qa[kc2][e] = a0[e]; qa[kc2][8 + e] = a1[e]; }
  }

  wait_async_le32();        // K issues (first 32, in-order) retired
  __syncthreads();

  // ---- Phase 1: S = Q K^T (K from LDS) ----
  for (int nt = 0; nt < 32; ++nt) {
    v8f s = {};
#pragma unroll
    for (int kc2 = 0; kc2 < 2; ++kc2) {
      v16bf bfrag =
          *(const v16bf*)(Ksh + (nt * 16 + lm) * KSTR + kc2 * 32 + lh * 16);
      s = wmma_bf16(qa[kc2], bfrag, s);
    }
#pragma unroll
    for (int e = 0; e < 8; ++e) {
      int m = wave * 16 + e + lh * 8;
      Ssh[m * SSTR + nt * 16 + lm] = s[e];
    }
  }
  __syncthreads();

  // ---- Phase 2: softmax over 512 keys; 2 lanes per row ----
  {
    int r = wave * 16 + lm;
    float* row = Ssh + r * SSTR + lh * 256;
    float mx = -3.0e38f;
    for (int j = 0; j < 256; ++j) mx = fmaxf(mx, row[j]);
    mx = fmaxf(mx, __shfl_xor(mx, 16, 32));
    float sum = 0.f;
    for (int j = 0; j < 256; ++j) {
      float ev = __expf(row[j] - mx);
      row[j] = ev;
      sum += ev;
    }
    sum += __shfl_xor(sum, 16, 32);
    if (lh == 0) rInv[r] = 1.0f / sum;
  }
  wait_async_0();           // V tile resident
  __syncthreads();

  // ---- Phase 3: O = P V (P from LDS f32->bf16, V from LDS) ----
  v8f o[4] = {};
  for (int kc = 0; kc < LGRP / 32; ++kc) {
    const float* pr = Ssh + (wave * 16 + lm) * SSTR + kc * 32;
    v16bf a;
#pragma unroll
    for (int e = 0; e < 8; ++e) {
      a[e]     = (bf16_t)pr[lh * 16 + e];
      a[8 + e] = (bf16_t)pr[lh * 16 + 8 + e];
    }
#pragma unroll
    for (int t = 0; t < 4; ++t) {
      v16bf bfrag =
          *(const v16bf*)(Vsh + (t * 16 + lm) * VSTR + kc * 32 + lh * 16);
      o[t] = wmma_bf16(a, bfrag, o[t]);
    }
  }

#pragma unroll
  for (int t = 0; t < 4; ++t) {
#pragma unroll
    for (int e = 0; e < 8; ++e) {
      int mloc = wave * 16 + e + lh * 8;
      float ov = o[t][e] * rInv[mloc];
      Oh[(rowbase + qc * 64 + mloc) * D_MODEL + h * DHD + t * 16 + lm] =
          (bf16_t)ov;
    }
  }
}

// ---------------------------------------------------------------------------
// Kernel 4: output projection  out[r,n] = sum_k Oh[r,k]*Wout[n,k] + b[n]
// ---------------------------------------------------------------------------
__global__ void out_gemm_kernel(const bf16_t* __restrict__ Oh,
                                const bf16_t* __restrict__ wo,
                                const float* __restrict__ bias,
                                float* __restrict__ out) {
  const int NGRP = D_MODEL / 64;   // 16
  int wid   = blockIdx.x * (blockDim.x >> 5) + (threadIdx.x >> 5);
  int mtile = wid / NGRP;
  int ngrp  = wid % NGRP;
  int lane  = threadIdx.x & 31;
  int lm = lane & 15, lh = lane >> 4;

  const bf16_t* arow = Oh + (size_t)(mtile * 16 + lm) * D_MODEL;
  v8f acc[4] = {};

  for (int kk = 0; kk < D_MODEL; kk += 32) {
    __builtin_prefetch(arow + kk + 256, 0, 0);
    v16bf a;
    v8bf a0 = *(const v8bf*)(arow + kk + lh * 16);
    v8bf a1 = *(const v8bf*)(arow + kk + lh * 16 + 8);
#pragma unroll
    for (int e = 0; e < 8; ++e) { a[e] = a0[e]; a[8 + e] = a1[e]; }
#pragma unroll
    for (int t = 0; t < 4; ++t) {
      int n = ngrp * 64 + t * 16 + lm;
      v16bf bfrag = *(const v16bf*)(wo + (size_t)n * D_MODEL + kk + lh * 16);
      acc[t] = wmma_bf16(a, bfrag, acc[t]);
    }
  }

#pragma unroll
  for (int t = 0; t < 4; ++t) {
    int n = ngrp * 64 + t * 16 + lm;
    float bn = bias[n];
#pragma unroll
    for (int e = 0; e < 8; ++e) {
      out[(size_t)(mtile * 16 + e + lh * 8) * D_MODEL + n] = acc[t][e] + bn;
    }
  }
}

// ---------------------------------------------------------------------------
extern "C" void kernel_launch(void* const* d_in, const int* in_sizes, int n_in,
                              void* d_out, int out_size, void* d_ws,
                              size_t ws_size, hipStream_t stream) {
  const float* x     = (const float*)d_in[0];
  const float* w_qkv = (const float*)d_in[1];
  const float* w_out = (const float*)d_in[2];
  const float* b_out = (const float*)d_in[3];
  float* out = (float*)d_out;

  char* ws = (char*)d_ws;
  bf16_t* Wqkvh = (bf16_t*)ws;  ws += (size_t)3 * D_MODEL * D_MODEL * 2;
  bf16_t* Wouth = (bf16_t*)ws;  ws += (size_t)D_MODEL * D_MODEL * 2;
  bf16_t* Qh    = (bf16_t*)ws;  ws += (size_t)NROWS * D_MODEL * 2;
  bf16_t* Kh    = (bf16_t*)ws;  ws += (size_t)NROWS * D_MODEL * 2;
  bf16_t* Vt    = (bf16_t*)ws;  ws += (size_t)NROWS * D_MODEL * 2;
  bf16_t* Oh    = (bf16_t*)ws;  ws += (size_t)NROWS * D_MODEL * 2;

  {
    int n = 3 * D_MODEL * D_MODEL;
    cvt_weights_kernel<<<(n + 255) / 256, 256, 0, stream>>>(w_qkv, w_out,
                                                            Wqkvh, Wouth);
  }
  {
    int jobs = (NROWS / 16) * ((3 * D_MODEL) / 64);   // 49152 waves
    qkv_gemm_kernel<<<jobs / 4, 128, 0, stream>>>(x, Wqkvh, Qh, Kh, Vt);
  }
  {
    int blocks = NBATCH * NH * NGROUP * (LGRP / 64);  // 4096
    attn_kernel<<<blocks, 128, SMEM_ATTN, stream>>>(Qh, Kh, Vt, Oh);
  }
  {
    int jobs = (NROWS / 16) * (D_MODEL / 64);         // 16384 waves
    out_gemm_kernel<<<jobs / 4, 128, 0, stream>>>(Oh, Wouth, b_out, out);
  }
}
